// SlideGraphArch_13065290514455
// MI455X (gfx1250) — compile-verified, matrix operand order checked by reading, not run on hardware
//
#include <hip/hip_runtime.h>
#include <hip/hip_bf16.h>

typedef __attribute__((ext_vector_type(16))) _Float16 v16h;
typedef __attribute__((ext_vector_type(8)))  _Float16 v8h;
typedef __attribute__((ext_vector_type(8)))  float    v8f;

#define N_NODES_C  100000
#define N_EDGES_C  600000
#define N_GRAPHS_C 8
#define D_IN_C     256
#define H_C        128
#define BN_EPS_C   1e-5f

// ---------- float <-> monotonic int key (for atomicMax-based segment max) ----
__device__ __forceinline__ int f32_key(float f) {
    int i = __float_as_int(f);
    return (i < 0) ? (i ^ 0x7FFFFFFF) : i;
}
__device__ __forceinline__ float key_f32(int k) {
    int i = (k < 0) ? (k ^ 0x7FFFFFFF) : k;
    return __int_as_float(i);
}

// ---------- prep: transpose+convert weights to f16, fold BN, init scratch ----
__global__ void __launch_bounds__(256)
prep_kernel(const float* __restrict__ w_first, const float* __restrict__ w_conv,
            const float* __restrict__ bn_first, const float* __restrict__ bn_conv,
            _Float16* __restrict__ wT0, _Float16* __restrict__ wT1,
            _Float16* __restrict__ wT2, float* __restrict__ bn_s,
            float* __restrict__ bn_t, int* __restrict__ seg) {
    const int tid = threadIdx.x;
    // wT0[n][k] = w_first[k][n]   (128 x 256)
    for (int idx = tid; idx < H_C * D_IN_C; idx += 256) {
        int n = idx / D_IN_C, k = idx - n * D_IN_C;
        wT0[idx] = (_Float16)w_first[k * H_C + n];
    }
    // wT1/wT2[n][k] = w_conv[l][k][n]   (128 x 128)
    for (int idx = tid; idx < H_C * H_C; idx += 256) {
        int n = idx / H_C, k = idx - n * H_C;
        wT1[idx] = (_Float16)w_conv[0 * H_C * H_C + k * H_C + n];
        wT2[idx] = (_Float16)w_conv[1 * H_C * H_C + k * H_C + n];
    }
    if (tid < H_C) {
        {   // layer 0: bn_first rows (gamma,beta,mean,var)
            float g = bn_first[0 * H_C + tid], b = bn_first[1 * H_C + tid];
            float m = bn_first[2 * H_C + tid], v = bn_first[3 * H_C + tid];
            float s = g * rsqrtf(v + BN_EPS_C);
            bn_s[tid] = s; bn_t[tid] = b - m * s;
        }
        for (int l = 0; l < 2; ++l) {
            const float* p = bn_conv + (size_t)l * 4 * H_C;
            float g = p[0 * H_C + tid], b = p[1 * H_C + tid];
            float m = p[2 * H_C + tid], v = p[3 * H_C + tid];
            float s = g * rsqrtf(v + BN_EPS_C);
            bn_s[(l + 1) * H_C + tid] = s;
            bn_t[(l + 1) * H_C + tid] = b - m * s;
        }
    }
    if (tid < 3 * N_GRAPHS_C) seg[tid] = (int)0x80000000;  // -inf key
}

// ---------- GEMM1: h = relu(bn0(x @ w_first + b_first)), K = 256 -------------
__global__ void __launch_bounds__(256)
gemm_first_kernel(const float* __restrict__ x, const _Float16* __restrict__ wT,
                  const float* __restrict__ bias, const float* __restrict__ bn_s,
                  const float* __restrict__ bn_t, float* __restrict__ h_out) {
    const int lane = threadIdx.x & 31;
    const int wave = threadIdx.x >> 5;
    const int hl   = lane >> 4;          // half-wave: 0 or 1
    const int n    = lane & 15;
    const int koff = hl * 8;
    const int rowBase = blockIdx.x * 16;
    const int col  = wave * 16 + n;

    const float*    xr = x  + (size_t)(rowBase + n) * D_IN_C;
    const _Float16* wr = wT + (size_t)col * D_IN_C;

    v8f c = {};
#pragma unroll
    for (int k0 = 0; k0 < D_IN_C; k0 += 32) {
        v16h a, b;
        float4 f0 = *(const float4*)(xr + k0 + koff);
        float4 f1 = *(const float4*)(xr + k0 + koff + 4);
        float4 f2 = *(const float4*)(xr + k0 + 16 + koff);
        float4 f3 = *(const float4*)(xr + k0 + 16 + koff + 4);
        a[0]=(_Float16)f0.x; a[1]=(_Float16)f0.y; a[2]=(_Float16)f0.z; a[3]=(_Float16)f0.w;
        a[4]=(_Float16)f1.x; a[5]=(_Float16)f1.y; a[6]=(_Float16)f1.z; a[7]=(_Float16)f1.w;
        a[8]=(_Float16)f2.x; a[9]=(_Float16)f2.y; a[10]=(_Float16)f2.z; a[11]=(_Float16)f2.w;
        a[12]=(_Float16)f3.x; a[13]=(_Float16)f3.y; a[14]=(_Float16)f3.z; a[15]=(_Float16)f3.w;
        v8h blo = *(const v8h*)(wr + k0 + koff);
        v8h bhi = *(const v8h*)(wr + k0 + 16 + koff);
#pragma unroll
        for (int j = 0; j < 8; ++j) { b[j] = blo[j]; b[8 + j] = bhi[j]; }
        c = __builtin_amdgcn_wmma_f32_16x16x32_f16(false, a, false, b,
                                                   (short)0, c, false, false);
    }
    const float bi = bias[col], s = bn_s[col], t = bn_t[col];
#pragma unroll
    for (int r = 0; r < 8; ++r) {
        int row = rowBase + r + hl * 8;
        float v = (c[r] + bi) * s + t;
        v = v > 0.f ? v : 0.f;
        h_out[(size_t)row * H_C + col] = v;
    }
}

// ---------- GIN conv GEMM: h = relu(bn((h+agg) @ w + b)), K = 128, in place --
__global__ void __launch_bounds__(256)
gemm_conv_kernel(const float* h_in, const float* __restrict__ agg,
                 const _Float16* __restrict__ wT, const float* __restrict__ bias,
                 const float* __restrict__ bn_s, const float* __restrict__ bn_t,
                 float* h_out) {
    const int lane = threadIdx.x & 31;
    const int wave = threadIdx.x >> 5;
    const int hl   = lane >> 4;
    const int n    = lane & 15;
    const int koff = hl * 8;
    const int rowBase = blockIdx.x * 16;
    const int col  = wave * 16 + n;

    const float*    hr = h_in + (size_t)(rowBase + n) * H_C;
    const float*    ar = agg  + (size_t)(rowBase + n) * H_C;
    const _Float16* wr = wT   + (size_t)col * H_C;

    v8f c = {};
#pragma unroll
    for (int k0 = 0; k0 < H_C; k0 += 32) {
        v16h a, b;
        float4 h0 = *(const float4*)(hr + k0 + koff);
        float4 h1 = *(const float4*)(hr + k0 + koff + 4);
        float4 h2 = *(const float4*)(hr + k0 + 16 + koff);
        float4 h3 = *(const float4*)(hr + k0 + 16 + koff + 4);
        float4 g0 = *(const float4*)(ar + k0 + koff);
        float4 g1 = *(const float4*)(ar + k0 + koff + 4);
        float4 g2 = *(const float4*)(ar + k0 + 16 + koff);
        float4 g3 = *(const float4*)(ar + k0 + 16 + koff + 4);
        a[0]=(_Float16)(h0.x+g0.x); a[1]=(_Float16)(h0.y+g0.y);
        a[2]=(_Float16)(h0.z+g0.z); a[3]=(_Float16)(h0.w+g0.w);
        a[4]=(_Float16)(h1.x+g1.x); a[5]=(_Float16)(h1.y+g1.y);
        a[6]=(_Float16)(h1.z+g1.z); a[7]=(_Float16)(h1.w+g1.w);
        a[8]=(_Float16)(h2.x+g2.x); a[9]=(_Float16)(h2.y+g2.y);
        a[10]=(_Float16)(h2.z+g2.z); a[11]=(_Float16)(h2.w+g2.w);
        a[12]=(_Float16)(h3.x+g3.x); a[13]=(_Float16)(h3.y+g3.y);
        a[14]=(_Float16)(h3.z+g3.z); a[15]=(_Float16)(h3.w+g3.w);
        v8h blo = *(const v8h*)(wr + k0 + koff);
        v8h bhi = *(const v8h*)(wr + k0 + 16 + koff);
#pragma unroll
        for (int j = 0; j < 8; ++j) { b[j] = blo[j]; b[8 + j] = bhi[j]; }
        c = __builtin_amdgcn_wmma_f32_16x16x32_f16(false, a, false, b,
                                                   (short)0, c, false, false);
    }
    const float bi = bias[col], s = bn_s[col], t = bn_t[col];
#pragma unroll
    for (int r = 0; r < 8; ++r) {
        int row = rowBase + r + hl * 8;
        float v = (c[r] + bi) * s + t;
        v = v > 0.f ? v : 0.f;
        h_out[(size_t)row * H_C + col] = v;
    }
}

// ---------- edge scatter: agg[dst] += h[src], one edge per wave --------------
__global__ void __launch_bounds__(256)
gather_kernel(const float* __restrict__ h, const int* __restrict__ src,
              const int* __restrict__ dst, float* agg) {
    const int wave = threadIdx.x >> 5;
    const int lane = threadIdx.x & 31;
    const int e = blockIdx.x * 8 + wave;
    if (e >= N_EDGES_C) return;
    const int s = src[e], d = dst[e];
    float4 v = *(const float4*)(h + (size_t)s * H_C + lane * 4);
    float* p = agg + (size_t)d * H_C + lane * 4;
    unsafeAtomicAdd(p + 0, v.x);
    unsafeAtomicAdd(p + 1, v.y);
    unsafeAtomicAdd(p + 2, v.z);
    unsafeAtomicAdd(p + 3, v.w);
}

// ---------- node head: np = h . w_lin + b; node_pred (+)= np; seg max --------
__global__ void __launch_bounds__(256)
node_head_kernel(const float* __restrict__ h, const float* __restrict__ wl,
                 const float* __restrict__ bl, const int* __restrict__ batch,
                 float* node_pred, int* seg, int accumulate) {
    const int wave = threadIdx.x >> 5;
    const int lane = threadIdx.x & 31;
    const int node = blockIdx.x * 8 + wave;
    if (node >= N_NODES_C) return;
    float4 hv = *(const float4*)(h + (size_t)node * H_C + lane * 4);
    float4 wv = *(const float4*)(wl + lane * 4);
    float d = hv.x * wv.x + hv.y * wv.y + hv.z * wv.z + hv.w * wv.w;
#pragma unroll
    for (int off = 16; off > 0; off >>= 1) d += __shfl_xor(d, off, 32);
    if (lane == 0) {
        float np = d + bl[0];
        if (accumulate) node_pred[node] += np;
        else            node_pred[node]  = np;
        atomicMax(seg + batch[node], f32_key(np));
    }
}

// ---------- finalize: wsi[g] = sum over layers of decoded segment maxes ------
__global__ void finalize_kernel(const int* __restrict__ seg, float* __restrict__ wsi) {
    int g = threadIdx.x;
    if (g < N_GRAPHS_C)
        wsi[g] = key_f32(seg[g]) + key_f32(seg[N_GRAPHS_C + g]) +
                 key_f32(seg[2 * N_GRAPHS_C + g]);
}

extern "C" void kernel_launch(void* const* d_in, const int* in_sizes, int n_in,
                              void* d_out, int out_size, void* d_ws, size_t ws_size,
                              hipStream_t stream) {
    const float* x        = (const float*)d_in[0];
    const int*   edge     = (const int*)d_in[1];   // [2][600000]: src then dst
    const int*   batch    = (const int*)d_in[2];
    const float* w_first  = (const float*)d_in[3];
    const float* b_first  = (const float*)d_in[4];
    const float* bn_first = (const float*)d_in[5];
    const float* w_conv   = (const float*)d_in[6];
    const float* b_conv   = (const float*)d_in[7];
    const float* bn_conv  = (const float*)d_in[8];
    const float* w_lin    = (const float*)d_in[9];  // [3][128]
    const float* b_lin    = (const float*)d_in[10]; // [3]

    char* ws = (char*)d_ws;
    const size_t HB = (size_t)N_NODES_C * H_C * sizeof(float);  // 51,200,000
    float*    h    = (float*)(ws);
    float*    agg  = (float*)(ws + HB);
    _Float16* wT0  = (_Float16*)(ws + 2 * HB);                       // 64 KB
    _Float16* wT1  = (_Float16*)(ws + 2 * HB + 65536);               // 32 KB
    _Float16* wT2  = (_Float16*)(ws + 2 * HB + 65536 + 32768);       // 32 KB
    float*    bn_s = (float*)(ws + 2 * HB + 131072);                 // 3*128 f32
    float*    bn_t = (float*)(ws + 2 * HB + 131072 + 1536);
    int*      seg  = (int*)(ws + 2 * HB + 131072 + 3072);            // 3*8 ints

    float* wsi       = (float*)d_out;               // [8]
    float* node_pred = (float*)d_out + N_GRAPHS_C;  // [100000]

    prep_kernel<<<1, 256, 0, stream>>>(w_first, w_conv, bn_first, bn_conv,
                                       wT0, wT1, wT2, bn_s, bn_t, seg);

    gemm_first_kernel<<<N_NODES_C / 16, 256, 0, stream>>>(
        x, wT0, b_first, bn_s, bn_t, h);

    node_head_kernel<<<(N_NODES_C + 7) / 8, 256, 0, stream>>>(
        h, w_lin, b_lin, batch, node_pred, seg, 0);

    for (int l = 0; l < 2; ++l) {
        hipMemsetAsync(agg, 0, HB, stream);
        gather_kernel<<<(N_EDGES_C + 7) / 8, 256, 0, stream>>>(
            h, edge, edge + N_EDGES_C, agg);
        const _Float16* wT = (l == 0) ? wT1 : wT2;
        gemm_conv_kernel<<<N_NODES_C / 16, 256, 0, stream>>>(
            h, agg, wT, b_conv + (size_t)l * H_C,
            bn_s + (size_t)(l + 1) * H_C, bn_t + (size_t)(l + 1) * H_C, h);
        node_head_kernel<<<(N_NODES_C + 7) / 8, 256, 0, stream>>>(
            h, w_lin + (size_t)(l + 1) * H_C, b_lin + (l + 1), batch,
            node_pred, seg + (l + 1) * N_GRAPHS_C, 1);
    }

    finalize_kernel<<<1, 32, 0, stream>>>(seg, wsi);
}
